// MoEBlock_53721450938817
// MI455X (gfx1250) — compile-verified, matrix operand order checked by reading, not run on hardware
//
#include <hip/hip_runtime.h>
#include <math.h>
#include <stdint.h>

typedef __attribute__((ext_vector_type(16))) __bf16 v16bf;
typedef __attribute__((ext_vector_type(8)))  float  v8f;
typedef __bf16 bf16_t;

#define TTOK 8192
#define DDIM 1024
#define IDIM 4096
#define ENUM 8
#define TOPK 2
#define BM 128
#define BN 64
#define BK 64
#define LDP (BK + 8)   // padded LDS row stride (elements); keeps 16B alignment
#define NT 256

#define ABUF_BYTES (BM * LDP * 2)
#define BBUF_BYTES (BN * LDP * 2)

// ---------------------------------------------------------------------------
// CDNA5 async copy: GLOBAL_LOAD_ASYNC_TO_LDS_B128 (tracked by ASYNCcnt).
// vdst = per-lane LDS byte address, vaddr = 64-bit global byte address.
// ---------------------------------------------------------------------------
__device__ __forceinline__ void async_cp16(uint32_t lds_off, uint64_t gaddr) {
  asm volatile("global_load_async_to_lds_b128 %0, %1, off"
               :: "v"(lds_off), "v"(gaddr)
               : "memory");
}
__device__ __forceinline__ void async_wait0() {
  asm volatile("s_wait_asynccnt 0x0" ::: "memory");
}
__device__ __forceinline__ uint32_t lds_addr(const void* p) {
  // Flat pointers to LDS carry the LDS byte offset in the low 32 bits
  // (ISA 10.2: upper address bits are ignored).
  return (uint32_t)(uintptr_t)p;
}

// ---------------------------------------------------------------------------
// Fragment load from LDS following CDNA5 ISA 16-bit A-matrix 16x32 layout:
// lane half selects K-group (+8), VGPR v covers K = 16*(v>>2) + 8*half + 2*(v&3).
// B (KxN) uses the mirrored mapping with lane%16 as the N index.
// ---------------------------------------------------------------------------
__device__ __forceinline__ v16bf frag_ld(const bf16_t* __restrict__ base, int ld,
                                         int kbase, int lane) {
  const int half = lane >> 4;
  const bf16_t* rp = base + (lane & 15) * ld;
  v16bf f;
#pragma unroll
  for (int v = 0; v < 8; ++v) {
    int kb = kbase + ((v >> 2) << 4) + (half << 3) + ((v & 3) << 1);
    f[2 * v]     = rp[kb];
    f[2 * v + 1] = rp[kb + 1];
  }
  return f;
}

// ---------------------------------------------------------------------------
// fp32 -> bf16 conversion (grid-stride). Weights become L2-resident (151MB<192MB).
// ---------------------------------------------------------------------------
__global__ void cvt_bf16_kernel(const float* __restrict__ src, bf16_t* __restrict__ dst, long n) {
  long i = (long)blockIdx.x * blockDim.x + threadIdx.x;
  long stride = (long)gridDim.x * blockDim.x;
  for (; i < n; i += stride) dst[i] = (bf16_t)src[i];
}

__global__ void init_kernel(int* __restrict__ cnt) {
  int i = threadIdx.x;
  if (i < ENUM) cnt[i] = 0;
}

// ---------------------------------------------------------------------------
// Gating: logits -> softmax over 8 experts -> top-2 (weights are softmax probs,
// not renormalized, matching the reference). One thread per token.
// ---------------------------------------------------------------------------
__global__ void gate_kernel(const float* __restrict__ x, const float* __restrict__ gw,
                            int* __restrict__ cnt, int* __restrict__ tokExp,
                            float* __restrict__ tokW) {
  int t = blockIdx.x * blockDim.x + threadIdx.x;
  if (t >= TTOK) return;
  const float* xr = x + (long)t * DDIM;
  float lg[ENUM];
#pragma unroll
  for (int e = 0; e < ENUM; ++e) lg[e] = 0.0f;
  for (int d = 0; d < DDIM; d += 4) {
    float4 xv = *(const float4*)(xr + d);
#pragma unroll
    for (int e = 0; e < ENUM; ++e) {
      float4 wv = *(const float4*)(gw + (long)e * DDIM + d);
      lg[e] += xv.x * wv.x + xv.y * wv.y + xv.z * wv.z + xv.w * wv.w;
    }
  }
  float mx = lg[0];
#pragma unroll
  for (int e = 1; e < ENUM; ++e) mx = fmaxf(mx, lg[e]);
  float p[ENUM];
  float den = 0.0f;
#pragma unroll
  for (int e = 0; e < ENUM; ++e) { p[e] = expf(lg[e] - mx); den += p[e]; }
  float inv = 1.0f / den;
  int i0 = 0;
#pragma unroll
  for (int e = 1; e < ENUM; ++e) if (p[e] > p[i0]) i0 = e;
  int i1 = (i0 == 0) ? 1 : 0;
#pragma unroll
  for (int e = 0; e < ENUM; ++e) if (e != i0 && p[e] > p[i1]) i1 = e;
  tokExp[2 * t]     = i0;  tokW[2 * t]     = p[i0] * inv;
  tokExp[2 * t + 1] = i1;  tokW[2 * t + 1] = p[i1] * inv;
  atomicAdd(&cnt[i0], 1);
  atomicAdd(&cnt[i1], 1);
}

__global__ void scan_kernel(const int* __restrict__ cnt, int* __restrict__ off,
                            int* __restrict__ cur) {
  if (threadIdx.x == 0 && blockIdx.x == 0) {
    int s = 0;
    for (int e = 0; e < ENUM; ++e) { off[e] = s; cur[e] = s; s += cnt[e]; }
  }
}

__global__ void scatter_kernel(const int* __restrict__ tokExp, const float* __restrict__ tokW,
                               int* __restrict__ cur, int* __restrict__ rowTok,
                               float* __restrict__ rowW) {
  int t = blockIdx.x * blockDim.x + threadIdx.x;
  if (t >= TTOK) return;
#pragma unroll
  for (int k = 0; k < TOPK; ++k) {
    int e = tokExp[2 * t + k];
    int pos = atomicAdd(&cur[e], 1);
    rowTok[pos] = t;
    rowW[pos] = tokW[2 * t + k];
  }
}

// ---------------------------------------------------------------------------
// FFN1 grouped GEMM: H = GELU(Xg @ W1^T + b1) for gathered rows of expert e.
// Double-buffered async-to-LDS pipeline; 8 waves x (2x2) v_wmma_f32_16x16x32_bf16.
// ---------------------------------------------------------------------------
__global__ __launch_bounds__(NT) void ffn1_kernel(
    const bf16_t* __restrict__ X, const bf16_t* __restrict__ W1,
    const float* __restrict__ B1, long wstride, long bstride,
    const int* __restrict__ rowTok, const int* __restrict__ cnt,
    const int* __restrict__ off, bf16_t* __restrict__ H, int shared_mode) {
  const int e = blockIdx.z;
  int rows, rbase;
  if (shared_mode) { rows = TTOK; rbase = 0; }
  else             { rows = cnt[e]; rbase = off[e]; }
  const int m0 = blockIdx.y * BM;
  if (m0 >= rows) return;
  const int n0 = blockIdx.x * BN;
  const bf16_t* W = W1 + (long)e * wstride;
  const float* bias = B1 + (long)e * bstride;

  __shared__ __align__(16) bf16_t As[2][BM][LDP];
  __shared__ __align__(16) bf16_t Bs[2][BN][LDP];

  const int tid = threadIdx.x;
  const int lane = tid & 31, wid = tid >> 5;
  const int wm = (wid & 3) * 32, wn = (wid >> 2) * 32;

  // Precompute per-thread staging addresses (token gather hoisted out of K loop).
  uint64_t aSrc[4];  uint32_t aDst[4];  bool aVal[4];
#pragma unroll
  for (int j = 0; j < 4; ++j) {
    int vec = tid + j * NT;           // 0..1023 : 128 rows x 8 vectors
    int row = vec >> 3;
    int kc  = (vec & 7) * 8;
    int gr  = m0 + row;
    aVal[j] = (gr < rows);
    int tok = aVal[j] ? (shared_mode ? gr : rowTok[rbase + gr]) : 0;
    aSrc[j] = (uint64_t)(uintptr_t)(X + (long)tok * DDIM + kc);
    aDst[j] = lds_addr(&As[0][row][kc]);
    if (!aVal[j]) {  // zero-fill OOB rows once; async never writes them
      *(uint4*)&As[0][row][kc] = make_uint4(0u, 0u, 0u, 0u);
      *(uint4*)&As[1][row][kc] = make_uint4(0u, 0u, 0u, 0u);
    }
  }
  uint64_t bSrc[2];  uint32_t bDst[2];
#pragma unroll
  for (int j = 0; j < 2; ++j) {
    int vec = tid + j * NT;           // 0..511 : 64 cols x 8 vectors
    int n  = vec >> 3;
    int kc = (vec & 7) * 8;
    bSrc[j] = (uint64_t)(uintptr_t)(W + (long)(n0 + n) * DDIM + kc);
    bDst[j] = lds_addr(&Bs[0][n][kc]);
  }

  v8f acc[2][2] = {};
  const int nk = DDIM / BK;

  // Prefetch stage 0 into buffer 0.
#pragma unroll
  for (int j = 0; j < 4; ++j)
    if (aVal[j]) async_cp16(aDst[j], aSrc[j]);
#pragma unroll
  for (int j = 0; j < 2; ++j) async_cp16(bDst[j], bSrc[j]);
  async_wait0();
  __syncthreads();

  for (int kt = 0; kt < nk; ++kt) {
    const int buf = kt & 1;
    // Prefetch next stage into the other buffer while computing this one.
    if (kt + 1 < nk) {
      const uint32_t adst = (buf ^ 1) * ABUF_BYTES;
      const uint32_t bdst = (buf ^ 1) * BBUF_BYTES;
      const uint64_t koff = (uint64_t)(kt + 1) * BK * 2;
#pragma unroll
      for (int j = 0; j < 4; ++j)
        if (aVal[j]) async_cp16(aDst[j] + adst, aSrc[j] + koff);
#pragma unroll
      for (int j = 0; j < 2; ++j) async_cp16(bDst[j] + bdst, bSrc[j] + koff);
      if (kt + 2 < nk)  // prime L2 for the weight stream two stages ahead
        __builtin_prefetch((const char*)(uintptr_t)bSrc[0] + (uint64_t)(kt + 2) * BK * 2, 0, 1);
    }
#pragma unroll
    for (int ks = 0; ks < BK; ks += 32) {
      v16bf a0 = frag_ld(&As[buf][wm][0],      LDP, ks, lane);
      v16bf a1 = frag_ld(&As[buf][wm + 16][0], LDP, ks, lane);
      v16bf b0 = frag_ld(&Bs[buf][wn][0],      LDP, ks, lane);
      v16bf b1 = frag_ld(&Bs[buf][wn + 16][0], LDP, ks, lane);
      acc[0][0] = __builtin_amdgcn_wmma_f32_16x16x32_bf16(false, a0, false, b0, (short)0, acc[0][0], false, false);
      acc[0][1] = __builtin_amdgcn_wmma_f32_16x16x32_bf16(false, a0, false, b1, (short)0, acc[0][1], false, false);
      acc[1][0] = __builtin_amdgcn_wmma_f32_16x16x32_bf16(false, a1, false, b0, (short)0, acc[1][0], false, false);
      acc[1][1] = __builtin_amdgcn_wmma_f32_16x16x32_bf16(false, a1, false, b1, (short)0, acc[1][1], false, false);
    }
    async_wait0();
    __syncthreads();
  }

  const int half = lane >> 4, nl = lane & 15;
#pragma unroll
  for (int i = 0; i < 2; ++i)
#pragma unroll
    for (int jj = 0; jj < 2; ++jj)
#pragma unroll
      for (int r = 0; r < 8; ++r) {
        int m = wm + i * 16 + half * 8 + r;
        int n = wn + jj * 16 + nl;
        int gr = m0 + m;
        if (gr < rows) {
          float v = acc[i][jj][r] + bias[n0 + n];
          float g = 0.5f * v * (1.0f + erff(v * 0.70710678118654752f));  // exact GELU
          H[(long)(rbase + gr) * IDIM + (n0 + n)] = (bf16_t)g;
        }
      }
}

// ---------------------------------------------------------------------------
// FFN2 grouped GEMM: out += we * (H @ W2^T + b2), scattered by token index.
// Shared mode stores directly (initializes d_out); expert mode atomicAdds.
// ---------------------------------------------------------------------------
__global__ __launch_bounds__(NT) void ffn2_kernel(
    const bf16_t* __restrict__ H, const bf16_t* __restrict__ W2,
    const float* __restrict__ B2, long wstride, long bstride,
    const int* __restrict__ rowTok, const float* __restrict__ rowW,
    const int* __restrict__ cnt, const int* __restrict__ off,
    float* __restrict__ out, int shared_mode) {
  const int e = blockIdx.z;
  int rows, rbase;
  if (shared_mode) { rows = TTOK; rbase = 0; }
  else             { rows = cnt[e]; rbase = off[e]; }
  const int m0 = blockIdx.y * BM;
  if (m0 >= rows) return;
  const int n0 = blockIdx.x * BN;
  const bf16_t* W = W2 + (long)e * wstride;
  const float* bias = B2 + (long)e * bstride;

  __shared__ __align__(16) bf16_t As[2][BM][LDP];
  __shared__ __align__(16) bf16_t Bs[2][BN][LDP];

  const int tid = threadIdx.x;
  const int lane = tid & 31, wid = tid >> 5;
  const int wm = (wid & 3) * 32, wn = (wid >> 2) * 32;

  uint64_t aSrc[4];  uint32_t aDst[4];  bool aVal[4];
#pragma unroll
  for (int j = 0; j < 4; ++j) {
    int vec = tid + j * NT;
    int row = vec >> 3;
    int kc  = (vec & 7) * 8;
    int gr  = m0 + row;
    aVal[j] = (gr < rows);
    int rr  = aVal[j] ? (rbase + gr) : 0;
    aSrc[j] = (uint64_t)(uintptr_t)(H + (long)rr * IDIM + kc);
    aDst[j] = lds_addr(&As[0][row][kc]);
    if (!aVal[j]) {
      *(uint4*)&As[0][row][kc] = make_uint4(0u, 0u, 0u, 0u);
      *(uint4*)&As[1][row][kc] = make_uint4(0u, 0u, 0u, 0u);
    }
  }
  uint64_t bSrc[2];  uint32_t bDst[2];
#pragma unroll
  for (int j = 0; j < 2; ++j) {
    int vec = tid + j * NT;
    int n  = vec >> 3;
    int kc = (vec & 7) * 8;
    bSrc[j] = (uint64_t)(uintptr_t)(W + (long)(n0 + n) * IDIM + kc);
    bDst[j] = lds_addr(&Bs[0][n][kc]);
  }

  v8f acc[2][2] = {};
  const int nk = IDIM / BK;

#pragma unroll
  for (int j = 0; j < 4; ++j)
    if (aVal[j]) async_cp16(aDst[j], aSrc[j]);
#pragma unroll
  for (int j = 0; j < 2; ++j) async_cp16(bDst[j], bSrc[j]);
  async_wait0();
  __syncthreads();

  for (int kt = 0; kt < nk; ++kt) {
    const int buf = kt & 1;
    if (kt + 1 < nk) {
      const uint32_t adst = (buf ^ 1) * ABUF_BYTES;
      const uint32_t bdst = (buf ^ 1) * BBUF_BYTES;
      const uint64_t koff = (uint64_t)(kt + 1) * BK * 2;
#pragma unroll
      for (int j = 0; j < 4; ++j)
        if (aVal[j]) async_cp16(aDst[j] + adst, aSrc[j] + koff);
#pragma unroll
      for (int j = 0; j < 2; ++j) async_cp16(bDst[j] + bdst, bSrc[j] + koff);
      if (kt + 2 < nk)
        __builtin_prefetch((const char*)(uintptr_t)bSrc[0] + (uint64_t)(kt + 2) * BK * 2, 0, 1);
    }
#pragma unroll
    for (int ks = 0; ks < BK; ks += 32) {
      v16bf a0 = frag_ld(&As[buf][wm][0],      LDP, ks, lane);
      v16bf a1 = frag_ld(&As[buf][wm + 16][0], LDP, ks, lane);
      v16bf b0 = frag_ld(&Bs[buf][wn][0],      LDP, ks, lane);
      v16bf b1 = frag_ld(&Bs[buf][wn + 16][0], LDP, ks, lane);
      acc[0][0] = __builtin_amdgcn_wmma_f32_16x16x32_bf16(false, a0, false, b0, (short)0, acc[0][0], false, false);
      acc[0][1] = __builtin_amdgcn_wmma_f32_16x16x32_bf16(false, a0, false, b1, (short)0, acc[0][1], false, false);
      acc[1][0] = __builtin_amdgcn_wmma_f32_16x16x32_bf16(false, a1, false, b0, (short)0, acc[1][0], false, false);
      acc[1][1] = __builtin_amdgcn_wmma_f32_16x16x32_bf16(false, a1, false, b1, (short)0, acc[1][1], false, false);
    }
    async_wait0();
    __syncthreads();
  }

  const int half = lane >> 4, nl = lane & 15;
#pragma unroll
  for (int i = 0; i < 2; ++i)
#pragma unroll
    for (int jj = 0; jj < 2; ++jj)
#pragma unroll
      for (int r = 0; r < 8; ++r) {
        int m = wm + i * 16 + half * 8 + r;
        int n = wn + jj * 16 + nl;
        int gr = m0 + m;
        if (gr < rows) {
          float v = acc[i][jj][r] + bias[n0 + n];
          if (shared_mode) {
            out[(long)gr * DDIM + (n0 + n)] = v;            // initializes output
          } else {
            int tok = rowTok[rbase + gr];
            float w = rowW[rbase + gr];
            atomicAdd(&out[(long)tok * DDIM + (n0 + n)], v * w);
          }
        }
      }
}

// ---------------------------------------------------------------------------
extern "C" void kernel_launch(void* const* d_in, const int* in_sizes, int n_in,
                              void* d_out, int out_size, void* d_ws, size_t ws_size,
                              hipStream_t stream) {
  const float* x   = (const float*)d_in[0];   // [B,S,D]
  const float* gw  = (const float*)d_in[1];   // [E,D]
  const float* ew1 = (const float*)d_in[2];   // [E,I,D]
  const float* eb1 = (const float*)d_in[3];   // [E,I]
  const float* ew2 = (const float*)d_in[4];   // [E,D,I]
  const float* eb2 = (const float*)d_in[5];   // [E,D]
  const float* sw1 = (const float*)d_in[6];   // [I,D]
  const float* sb1 = (const float*)d_in[7];   // [I]
  const float* sw2 = (const float*)d_in[8];   // [D,I]
  const float* sb2 = (const float*)d_in[9];   // [D]
  float* out = (float*)d_out;
  char* ws = (char*)d_ws;

  size_t o = 0;
  auto alloc = [&](size_t bytes) -> size_t {
    size_t r = o;
    o += (bytes + 255) & ~(size_t)255;
    return r;
  };
  bf16_t* Xbf  = (bf16_t*)(ws + alloc((size_t)TTOK * DDIM * 2));
  bf16_t* EW1  = (bf16_t*)(ws + alloc((size_t)ENUM * IDIM * DDIM * 2));
  bf16_t* EW2  = (bf16_t*)(ws + alloc((size_t)ENUM * IDIM * DDIM * 2));
  bf16_t* SW1  = (bf16_t*)(ws + alloc((size_t)IDIM * DDIM * 2));
  bf16_t* SW2  = (bf16_t*)(ws + alloc((size_t)IDIM * DDIM * 2));
  bf16_t* Hbuf = (bf16_t*)(ws + alloc((size_t)TOPK * TTOK * IDIM * 2));
  int*    rowTok = (int*)  (ws + alloc((size_t)TOPK * TTOK * 4));
  float*  rowW   = (float*)(ws + alloc((size_t)TOPK * TTOK * 4));
  int*    cnt    = (int*)  (ws + alloc(ENUM * 4));
  int*    off    = (int*)  (ws + alloc(ENUM * 4));
  int*    cur    = (int*)  (ws + alloc(ENUM * 4));
  int*    tokExp = (int*)  (ws + alloc((size_t)TOPK * TTOK * 4));
  float*  tokW   = (float*)(ws + alloc((size_t)TOPK * TTOK * 4));

  init_kernel<<<1, 32, 0, stream>>>(cnt);

  // bf16 conversions: ~151MB of weights -> fits the 192MB L2 for the GEMMs.
  cvt_bf16_kernel<<<2048, 256, 0, stream>>>(x,   Xbf, (long)TTOK * DDIM);
  cvt_bf16_kernel<<<2048, 256, 0, stream>>>(ew1, EW1, (long)ENUM * IDIM * DDIM);
  cvt_bf16_kernel<<<2048, 256, 0, stream>>>(ew2, EW2, (long)ENUM * IDIM * DDIM);
  cvt_bf16_kernel<<<2048, 256, 0, stream>>>(sw1, SW1, (long)IDIM * DDIM);
  cvt_bf16_kernel<<<2048, 256, 0, stream>>>(sw2, SW2, (long)IDIM * DDIM);

  // Routing.
  gate_kernel<<<TTOK / 256, 256, 0, stream>>>(x, gw, cnt, tokExp, tokW);
  scan_kernel<<<1, 32, 0, stream>>>(cnt, off, cur);
  scatter_kernel<<<TTOK / 256, 256, 0, stream>>>(tokExp, tokW, cur, rowTok, rowW);

  // Shared expert (writes out, initializing it).
  ffn1_kernel<<<dim3(IDIM / BN, TTOK / BM, 1), NT, 0, stream>>>(
      Xbf, SW1, sb1, 0, 0, nullptr, nullptr, nullptr, Hbuf, 1);
  ffn2_kernel<<<dim3(DDIM / BN, TTOK / BM, 1), NT, 0, stream>>>(
      Hbuf, SW2, sb2, 0, 0, nullptr, nullptr, nullptr, nullptr, out, 1);

  // Routed experts: grouped GEMM over gathered token lists, atomicAdd scatter.
  ffn1_kernel<<<dim3(IDIM / BN, TTOK / BM, ENUM), NT, 0, stream>>>(
      Xbf, EW1, eb1, (long)IDIM * DDIM, IDIM, rowTok, cnt, off, Hbuf, 0);
  ffn2_kernel<<<dim3(DDIM / BN, TTOK / BM, ENUM), NT, 0, stream>>>(
      Hbuf, EW2, eb2, (long)DDIM * IDIM, DDIM, rowTok, rowW, cnt, off, out, 0);
}
